// MultiHeadAttention_64415919506023
// MI455X (gfx1250) — matrix operand order repaired and checked by reading, not compiled, not run on hardware
//
#include <hip/hip_runtime.h>
#include <hip/hip_bf16.h>

// ---------------------------------------------------------------------------
// MultiHeadAttention block for MI455X (gfx1250, wave32, WMMA)
//   B=2, S=2048, D=1024, H=16, D_H=64
// Pipeline:
//   1) gemm<0>: Q = (x@Wq^T+b)*0.125 -> f16 [B,H,S,64]  (attn scale folded in)
//   2) gemm<0>: K = x@Wk^T+b         -> f16 [B,H,S,64]
//   3) gemm<1>: V = x@Wv^T+b         -> f16 [B,H,64,S]  (transposed for PV)
//   4) flash attention (causal, analytic mask; softmax denominator computed
//      as a 5th WMMA tile against an all-ones B matrix) -> f32 ctx [B,S,D]
//   5) gemm<2>: out = ctx@Wff^T+b    -> f32 [B,S,D]
// ---------------------------------------------------------------------------

#define BB   2
#define SS   2048
#define DD   1024
#define HH   16
#define DH   64

typedef _Float16 h8  __attribute__((ext_vector_type(8)));
typedef _Float16 h16 __attribute__((ext_vector_type(16)));
typedef float    f8  __attribute__((ext_vector_type(8)));

__device__ __forceinline__ h16 cat8(h8 lo, h8 hi) {
  return __builtin_shufflevector(lo, hi, 0,1,2,3,4,5,6,7,8,9,10,11,12,13,14,15);
}

__device__ __forceinline__ f8 zero_f8() {
  f8 z = {0.f,0.f,0.f,0.f,0.f,0.f,0.f,0.f};
  return z;
}

__device__ __forceinline__ h16 ones_h16() {
  h16 o = {(_Float16)1.f,(_Float16)1.f,(_Float16)1.f,(_Float16)1.f,
           (_Float16)1.f,(_Float16)1.f,(_Float16)1.f,(_Float16)1.f,
           (_Float16)1.f,(_Float16)1.f,(_Float16)1.f,(_Float16)1.f,
           (_Float16)1.f,(_Float16)1.f,(_Float16)1.f,(_Float16)1.f};
  return o;
}

__device__ __forceinline__ f8 wmma_f16(h16 a, h16 b, f8 c) {
  // D = A(16x32 f16) * B(32x16 f16) + C(16x16 f32)
  return __builtin_amdgcn_wmma_f32_16x16x32_f16(
      /*neg_a=*/false, a, /*neg_b=*/false, b,
      /*c_mod=*/(short)0, c, /*reuse_a=*/false, /*reuse_b=*/false);
}

__device__ __forceinline__ void wave_lds_fence() {
#if __has_builtin(__builtin_amdgcn_wave_barrier)
  __builtin_amdgcn_wave_barrier();
#else
  asm volatile("" ::: "memory");
#endif
}

// xor-butterfly max across the 16 lanes of a half-wave group
__device__ __forceinline__ float red16_max(float v, int lane) {
#pragma unroll
  for (int m = 8; m >= 1; m >>= 1) {
    int idx = ((lane ^ m) & 31) << 2;
    float o = __int_as_float(__builtin_amdgcn_ds_bpermute(idx, __float_as_int(v)));
    v = fmaxf(v, o);
  }
  return v;
}

// ---------------------------------------------------------------------------
// GEMM: Out[m,n] = (sum_k A[m,k] * W[n,k] + bias[n]) * scale
//   A: f32 [4096,1024] row-major, W: f32 [1024,1024] row-major
//   MODE 0: f16 out, head-split  [B,H,S,64]
//   MODE 1: f16 out, transposed  [B,H,64,S]
//   MODE 2: f32 out, plain       [M,N]
// Block tile 64(M) x 128(N) x 64(K), 8 waves, each wave 32x32 (2x2 WMMA).
// ---------------------------------------------------------------------------
template <int MODE>
__global__ __launch_bounds__(256) void gemm_kernel(
    const float* __restrict__ A, const float* __restrict__ W,
    const float* __restrict__ bias, void* __restrict__ Out, float scale) {
  __shared__ _Float16 At[64][64];    //  8 KB
  __shared__ _Float16 Bt[128][64];   // 16 KB

  const int t    = threadIdx.x;
  const int lane = t & 31;
  const int wave = t >> 5;
  const int g    = lane >> 4;   // lane half-group (K-half selector)
  const int ln   = lane & 15;
  const int m0   = blockIdx.y * 64;
  const int n0   = blockIdx.x * 128;
  const int wmo  = (wave >> 2) * 32;   // wave M offset in tile
  const int wno  = (wave & 3) * 32;    // wave N offset in tile

  f8 acc[2][2];
#pragma unroll
  for (int i = 0; i < 2; ++i)
#pragma unroll
    for (int j = 0; j < 2; ++j) acc[i][j] = zero_f8();

  for (int k0 = 0; k0 < DD; k0 += 64) {
    // ---- stage A tile: 64 rows x 64 cols f32 -> f16 LDS (1024 float4) ----
#pragma unroll
    for (int i = 0; i < 4; ++i) {
      int e = t + i * 256;               // 0..1023
      int row = e >> 4, c4 = e & 15;
      const float4 v = *(const float4*)(A + (size_t)(m0 + row) * DD + k0 + c4 * 4);
      At[row][c4 * 4 + 0] = (_Float16)v.x;
      At[row][c4 * 4 + 1] = (_Float16)v.y;
      At[row][c4 * 4 + 2] = (_Float16)v.z;
      At[row][c4 * 4 + 3] = (_Float16)v.w;
    }
    // ---- stage B tile: 128 rows(N) x 64 cols(K) (2048 float4) ----
#pragma unroll
    for (int i = 0; i < 8; ++i) {
      int e = t + i * 256;               // 0..2047
      int row = e >> 4, c4 = e & 15;
      const float4 v = *(const float4*)(W + (size_t)(n0 + row) * DD + k0 + c4 * 4);
      Bt[row][c4 * 4 + 0] = (_Float16)v.x;
      Bt[row][c4 * 4 + 1] = (_Float16)v.y;
      Bt[row][c4 * 4 + 2] = (_Float16)v.z;
      Bt[row][c4 * 4 + 3] = (_Float16)v.w;
    }
    __syncthreads();

    if (k0 + 64 < DD) {  // prefetch next K-tile (global_prefetch_b8)
      __builtin_prefetch(A + (size_t)(m0 + (t & 63)) * DD + (k0 + 64), 0, 1);
      __builtin_prefetch(W + (size_t)(n0 + (t & 127)) * DD + (k0 + 64), 0, 1);
    }

    // ---- two 32-wide k-steps out of LDS ----
#pragma unroll
    for (int kk = 0; kk < 2; ++kk) {
      h16 af[2], bf[2];
#pragma unroll
      for (int mt = 0; mt < 2; ++mt) {
        const _Float16* p = &At[wmo + mt * 16 + ln][kk * 32];
        af[mt] = cat8(*(const h8*)(p + g * 8), *(const h8*)(p + 16 + g * 8));
      }
#pragma unroll
      for (int nt = 0; nt < 2; ++nt) {
        const _Float16* p = &Bt[wno + nt * 16 + ln][kk * 32];
        bf[nt] = cat8(*(const h8*)(p + g * 8), *(const h8*)(p + 16 + g * 8));
      }
#pragma unroll
      for (int mt = 0; mt < 2; ++mt)
#pragma unroll
        for (int nt = 0; nt < 2; ++nt)
          acc[mt][nt] = wmma_f16(af[mt], bf[nt], acc[mt][nt]);
    }
    __syncthreads();
  }

  // ---- epilogue: C layout row = r + 8*g, col = ln ----
#pragma unroll
  for (int mt = 0; mt < 2; ++mt) {
#pragma unroll
    for (int nt = 0; nt < 2; ++nt) {
      const int nglob = n0 + wno + nt * 16 + ln;
      const float bv = bias[nglob];
#pragma unroll
      for (int r = 0; r < 8; ++r) {
        const int mglob = m0 + wmo + mt * 16 + 8 * g + r;
        const float val = (acc[mt][nt][r] + bv) * scale;
        if (MODE == 2) {
          ((float*)Out)[(size_t)mglob * DD + nglob] = val;
        } else {
          const int b = mglob >> 11, s = mglob & (SS - 1);
          const int hh = nglob >> 6, d = nglob & (DH - 1);
          size_t idx;
          if (MODE == 0)
            idx = (((size_t)b * HH + hh) * SS + s) * DH + d;        // [B,H,S,64]
          else
            idx = (((size_t)b * HH + hh) * DH + d) * SS + s;        // [B,H,64,S]
          ((_Float16*)Out)[idx] = (_Float16)val;
        }
      }
    }
  }
}

// ---------------------------------------------------------------------------
// Flash attention (causal). One wave = 16 query rows; block = 8 waves = 128
// rows of one (b,h). Key loop split: [0, jdiag) fully-unmasked fast path,
// then exactly one masked diagonal block. Softmax denominator accumulated as
// a 5th WMMA tile (P x ones), so only the row-max needs a lane reduction.
// ---------------------------------------------------------------------------
__global__ __launch_bounds__(256) void attn_kernel(
    const _Float16* __restrict__ Q, const _Float16* __restrict__ K,
    const _Float16* __restrict__ Vt, float* __restrict__ Ctx) {
  __shared__ _Float16 Pbuf[8][16][32];   // per-wave P staging (C-layout -> A-layout)

  const int t    = threadIdx.x;
  const int lane = t & 31;
  const int wave = t >> 5;
  const int g    = lane >> 4;
  const int ln   = lane & 15;
  const int bh   = blockIdx.y;
  const int b    = bh >> 4;
  const int h    = bh & 15;
  const int qbase = blockIdx.x * 128 + wave * 16;

  const _Float16* Qp = Q  + (size_t)bh * SS * DH;
  const _Float16* Kp = K  + (size_t)bh * SS * DH;
  const _Float16* Vp = Vt + (size_t)bh * DH * SS;

  // Q fragments, two 32-wide d chunks (held for the whole key loop)
  h16 qa[2];
  {
    const _Float16* qrow = Qp + (size_t)(qbase + ln) * DH;
#pragma unroll
    for (int kk = 0; kk < 2; ++kk)
      qa[kk] = cat8(*(const h8*)(qrow + kk * 32 + g * 8),
                    *(const h8*)(qrow + kk * 32 + 16 + g * 8));
  }
  const h16 onesf = ones_h16();

  float mrow[8];
  f8 oacc[4];       // output columns
  f8 oaccL;         // softmax denominator (P x ones): every lane = row sum
#pragma unroll
  for (int r = 0; r < 8; ++r) mrow[r] = -1e30f;
#pragma unroll
  for (int dt = 0; dt < 4; ++dt) oacc[dt] = zero_f8();
  oaccL = zero_f8();

  const int jdiag = qbase >> 5;  // the single partially-masked diagonal block

  for (int j = 0; j <= jdiag; ++j) {
    const int kb = j * 32;
    const bool masked = (j == jdiag);

    // ---- scores: S = Q * K^T (two 16-key N tiles, two d k-steps) ----
    f8 sc[2];
#pragma unroll
    for (int nt = 0; nt < 2; ++nt) {
      sc[nt] = zero_f8();
      const _Float16* krow = Kp + (size_t)(kb + nt * 16 + ln) * DH;
#pragma unroll
      for (int kk = 0; kk < 2; ++kk) {
        h16 kf = cat8(*(const h8*)(krow + kk * 32 + g * 8),
                      *(const h8*)(krow + kk * 32 + 16 + g * 8));
        sc[nt] = wmma_f16(qa[kk], kf, sc[nt]);
      }
    }

    // ---- causal mask (diagonal block only) + block row max ----
    float bm[8];
#pragma unroll
    for (int r = 0; r < 8; ++r) bm[r] = -1e30f;
    if (masked) {
#pragma unroll
      for (int nt = 0; nt < 2; ++nt) {
        const int kcol = kb + nt * 16 + ln;
#pragma unroll
        for (int r = 0; r < 8; ++r) {
          const int qr = qbase + 8 * g + r;
          float v = sc[nt][r];
          v = (kcol > qr) ? -1e30f : v;
          sc[nt][r] = v;
          bm[r] = fmaxf(bm[r], v);
        }
      }
    } else {
#pragma unroll
      for (int nt = 0; nt < 2; ++nt)
#pragma unroll
        for (int r = 0; r < 8; ++r) bm[r] = fmaxf(bm[r], sc[nt][r]);
    }
#pragma unroll
    for (int r = 0; r < 8; ++r) bm[r] = red16_max(bm[r], lane);

    // ---- online softmax rescale + P to LDS (f16, A-fragment staging) ----
#pragma unroll
    for (int r = 0; r < 8; ++r) {
      const float mn = fmaxf(mrow[r], bm[r]);
      const float corr = __expf(mrow[r] - mn);
      mrow[r] = mn;
      oaccL[r] *= corr;
#pragma unroll
      for (int dt = 0; dt < 4; ++dt) oacc[dt][r] *= corr;
    }
#pragma unroll
    for (int nt = 0; nt < 2; ++nt) {
#pragma unroll
      for (int r = 0; r < 8; ++r) {
        const float p = __expf(sc[nt][r] - mrow[r]);
        Pbuf[wave][8 * g + r][nt * 16 + ln] = (_Float16)p;
      }
    }
    wave_lds_fence();   // DS ops are in-order within a wave; block reordering

    // ---- P fragment (A layout) ----
    const _Float16* prow = &Pbuf[wave][ln][0];
    h16 pa = cat8(*(const h8*)(prow + g * 8), *(const h8*)(prow + 16 + g * 8));

    // ---- denominator tile + O += P * V (Vt[d][s] contiguous in keys) ----
    oaccL = wmma_f16(pa, onesf, oaccL);
#pragma unroll
    for (int dt = 0; dt < 4; ++dt) {
      const _Float16* vrow = Vp + (size_t)(dt * 16 + ln) * SS + kb;
      h16 vf = cat8(*(const h8*)(vrow + g * 8), *(const h8*)(vrow + 16 + g * 8));
      oacc[dt] = wmma_f16(pa, vf, oacc[dt]);
    }
    wave_lds_fence();   // before next iteration overwrites Pbuf
  }

  // ---- normalize + write context [B,S,D] f32 ----
#pragma unroll
  for (int dt = 0; dt < 4; ++dt) {
#pragma unroll
    for (int r = 0; r < 8; ++r) {
      const int qr = qbase + 8 * g + r;
      const int d = dt * 16 + ln;
      Ctx[((size_t)b * SS + qr) * DD + h * DH + d] = oacc[dt][r] / oaccL[r];
    }
  }
}

// ---------------------------------------------------------------------------
extern "C" void kernel_launch(void* const* d_in, const int* in_sizes, int n_in,
                              void* d_out, int out_size, void* d_ws, size_t ws_size,
                              hipStream_t stream) {
  const float* query = (const float*)d_in[0];
  const float* key   = (const float*)d_in[1];
  const float* value = (const float*)d_in[2];
  // d_in[3] = mask: causal triu, applied analytically in attn_kernel (saves 33MB of reads)
  const float* wq = (const float*)d_in[4];  const float* bq = (const float*)d_in[5];
  const float* wk = (const float*)d_in[6];  const float* bk = (const float*)d_in[7];
  const float* wv = (const float*)d_in[8];  const float* bv = (const float*)d_in[9];
  const float* wff = (const float*)d_in[10]; const float* bff = (const float*)d_in[11];
  float* out = (float*)d_out;

  // workspace layout: Q f16 (8MB) | K f16 (8MB) | Vt f16 (8MB) | ctx f32 (16MB) = 40MB
  const size_t nElem = (size_t)BB * HH * SS * DH;   // 4,194,304
  _Float16* qh = (_Float16*)d_ws;
  _Float16* kh = qh + nElem;
  _Float16* vh = kh + nElem;
  float*   ctx = (float*)(vh + nElem);

  dim3 ggrid(DD / 128, (BB * SS) / 64);   // (8, 64)
  // attention scale 1/sqrt(64) folded into the Q projection (linear => exact)
  gemm_kernel<0><<<ggrid, 256, 0, stream>>>(query, wq, bq, (void*)qh, 0.125f);
  gemm_kernel<0><<<ggrid, 256, 0, stream>>>(key,   wk, bk, (void*)kh, 1.0f);
  gemm_kernel<1><<<ggrid, 256, 0, stream>>>(value, wv, bv, (void*)vh, 1.0f);

  attn_kernel<<<dim3(SS / 128, BB * HH), 256, 0, stream>>>(qh, kh, vh, ctx);

  gemm_kernel<2><<<ggrid, 256, 0, stream>>>(ctx, wff, bff, (void*)out, 1.0f);
}
